// GRUPrefixEncodingComponent_83683142795688
// MI455X (gfx1250) — compile-verified, matrix operand order checked by reading, not run on hardware
//
#include <hip/hip_runtime.h>

// Problem dims (match reference)
#define B_DIM   128
#define S_DIM   512
#define IN_DIMS 512
#define HID_    1024
#define GATES_  (3 * HID_)

// LDS row stride: 64B of K-chunk data + 16B pad => bank-conflict-free for
// 16-lane stride-ROW_BYTES fragment reads, and 16B-aligned for ds_load_b128.
#define ROW_BYTES 80

typedef __attribute__((ext_vector_type(16))) __bf16 v16bf;
typedef __attribute__((ext_vector_type(8)))  float  v8f;

union FragU { uint4 q[2]; v16bf v; };

#if defined(__has_builtin) && __has_builtin(__builtin_amdgcn_sched_barrier)
#define SCHED_FENCE() __builtin_amdgcn_sched_barrier(0)
#else
#define SCHED_FENCE() do {} while (0)
#endif

// --- gfx1250 async copy: global -> LDS, 16B per lane, tracked by ASYNCcnt ---
static __device__ __forceinline__ void async_b128(unsigned lds_addr, const __bf16* gptr) {
  asm volatile("global_load_async_to_lds_b128 %0, %1, off"
               :: "v"(lds_addr), "v"(gptr)
               : "memory");
}
static __device__ __forceinline__ void wait_async0() {
  asm volatile("s_wait_asynccnt 0x0" ::: "memory");
}

// --- CDNA5 WMMA bf16 fragment loaders -----------------------------------
// 16-bit A/B matrix layout (ISA 7.12.2): lane L: row/col = L%16, khalf = L/16.
// VGPR v holds k = (v/4)*16 + khalf*8 + (v%4)*2 (+0,+1): two contiguous 8-elem
// (16B) groups at k-local {khalf*8, 16+khalf*8}.
static __device__ __forceinline__ v16bf frag_global(const __bf16* __restrict__ rowbase,
                                                    int k0, int khalf) {
  FragU u;
  u.q[0] = *reinterpret_cast<const uint4*>(rowbase + k0 + khalf * 8);
  u.q[1] = *reinterpret_cast<const uint4*>(rowbase + k0 + 16 + khalf * 8);
  return u.v;
}
static __device__ __forceinline__ v16bf frag_lds(const char* rowbase, int khalf) {
  FragU u;
  u.q[0] = *reinterpret_cast<const uint4*>(rowbase + khalf * 16);
  u.q[1] = *reinterpret_cast<const uint4*>(rowbase + 32 + khalf * 16);
  return u.v;
}

// --- helpers ------------------------------------------------------------
__global__ void f32_to_bf16_kernel(const float* __restrict__ in,
                                   __bf16* __restrict__ out, long n) {
  long i = (long)blockIdx.x * blockDim.x + threadIdx.x;
  if (i < n) out[i] = (__bf16)in[i];
}

__global__ void zero_h_kernel(__bf16* __restrict__ hbf, float* __restrict__ hf, int n) {
  int i = blockIdx.x * blockDim.x + threadIdx.x;
  if (i < n) { hbf[i] = (__bf16)0.0f; hf[i] = 0.0f; }
}

// --- Phase 1: gi_all[m][g] = X[m][:] . W_ih[g][:] + b_ih[g] -------------
// M = B*S = 65536, N = 3072, K = 512.  Block = 8 waves (256 thr),
// block tile 128(M) x 64(N); wave tile 16 x 64.  B (w_ih rows n0..n0+63)
// is shared by all 8 waves -> staged in LDS via async copy, double-buffered.
__global__ __launch_bounds__(256) void gi_gemm_kernel(
    const __bf16* __restrict__ X, const __bf16* __restrict__ W,
    const float* __restrict__ bias, float* __restrict__ GI) {
  __shared__ uint4 smem[2 * 64 * ROW_BYTES / 16];
  char* sbase = (char*)smem;
  const unsigned lds0 = (unsigned)(size_t)(void*)smem;   // LDS byte address
  const unsigned bufstride = 64 * ROW_BYTES;

  const int tid   = threadIdx.x;
  const int lane  = tid & 31;
  const int wave  = tid >> 5;
  const int row   = lane & 15;
  const int khalf = lane >> 4;
  const int n0 = blockIdx.x * 64;
  const int m0 = blockIdx.y * 128 + wave * 16;

  // staging map: one b128 per thread per chunk (64 rows x 4 segs = 256)
  const int ld_row = tid >> 2;   // 0..63
  const int ld_seg = tid & 3;    // 0..3
  const __bf16* wrow = W + (long)(n0 + ld_row) * IN_DIMS + ld_seg * 8;
  const unsigned lds_w = lds0 + (unsigned)(ld_row * ROW_BYTES + ld_seg * 16);

  const __bf16* arow = X + (long)(m0 + row) * IN_DIMS;

  async_b128(lds_w, wrow);                 // preload chunk 0
  v8f acc[4] = {};

  for (int ks = 0; ks < IN_DIMS / 32; ++ks) {
    const int cur = ks & 1;
    wait_async0();
    __syncthreads();
    if (ks + 1 < IN_DIMS / 32)
      async_b128(lds_w + (unsigned)((cur ^ 1) * bufstride), wrow + (ks + 1) * 32);

    v16bf a = frag_global(arow, ks * 32, khalf);

    // batch all DS reads (one clause, one wait), then a WMMA burst;
    // sched fence keeps the 4 fragment live ranges overlapping -> distinct regs
    v16bf bfr[4];
#pragma unroll
    for (int t = 0; t < 4; ++t)
      bfr[t] = frag_lds(sbase + cur * bufstride + (t * 16 + row) * ROW_BYTES, khalf);
    SCHED_FENCE();
#pragma unroll
    for (int t = 0; t < 4; ++t)
      acc[t] = __builtin_amdgcn_wmma_f32_16x16x32_bf16(
          false, a, false, bfr[t], (short)0, acc[t], false, false);
  }

  // C/D layout: lane L: N = L%16; VGPR r: M = r + (L>=16 ? 8 : 0)
#pragma unroll
  for (int t = 0; t < 4; ++t) {
    const int n = n0 + t * 16 + row;
    const float bv = bias[n];
#pragma unroll
    for (int r = 0; r < 8; ++r) {
      const int m = m0 + khalf * 8 + r;
      GI[(long)m * GATES_ + n] = acc[t][r] + bv;
    }
  }
}

// --- Phase 2: one GRU step ----------------------------------------------
// gh = h_prev @ W_hh^T ; gates fused.  Block = 4 waves (128 thr),
// block tile 64(batch) x 64(hidden j); wave tile 16 x 64 x 3 gates.
// W_hh rows {q*H + j0 .. +63 | q=0..2} (192 rows) shared by all waves ->
// staged in LDS per 32-K chunk via async copy, double-buffered.
__global__ __launch_bounds__(128) void gru_step_kernel(
    const __bf16* __restrict__ Hbf, const float* __restrict__ Hf,
    const __bf16* __restrict__ Whh, const float* __restrict__ GI,
    const float* __restrict__ bhh,
    __bf16* __restrict__ HbfN, float* __restrict__ HfN,
    float* __restrict__ outH, float* __restrict__ outLast, int s) {
  __shared__ uint4 smem[2 * 192 * ROW_BYTES / 16];
  char* sbase = (char*)smem;
  const unsigned lds0 = (unsigned)(size_t)(void*)smem;
  const unsigned bufstride = 192 * ROW_BYTES;

  const int tid   = threadIdx.x;
  const int lane  = tid & 31;
  const int wave  = tid >> 5;
  const int row   = lane & 15;
  const int khalf = lane >> 4;
  const int j0 = blockIdx.x * 64;
  const int b0 = blockIdx.y * 64 + wave * 16;

  // staging map: 192 rows x 4 segs = 768 b128 per chunk / 128 thr = 6 each
  const __bf16* gld[6];
  unsigned lld[6];
#pragma unroll
  for (int i = 0; i < 6; ++i) {
    const int l = tid + 128 * i;
    const int r = l >> 2;          // 0..191
    const int seg = l & 3;
    const int grow = (r >> 6) * HID_ + j0 + (r & 63);   // gate*H + j0 + col
    gld[i] = Whh + (long)grow * HID_ + seg * 8;
    lld[i] = lds0 + (unsigned)(r * ROW_BYTES + seg * 16);
  }

  const __bf16* arow = Hbf + (long)(b0 + row) * HID_;

#pragma unroll
  for (int i = 0; i < 6; ++i) async_b128(lld[i], gld[i]);   // preload chunk 0

  v8f acc[3][4] = {};

  for (int ks = 0; ks < HID_ / 32; ++ks) {
    const int cur = ks & 1;
    wait_async0();
    __syncthreads();
    if (ks + 1 < HID_ / 32) {
      const unsigned boff = (unsigned)((cur ^ 1) * bufstride);
      const int koff = (ks + 1) * 32;
#pragma unroll
      for (int i = 0; i < 6; ++i) async_b128(lld[i] + boff, gld[i] + koff);
    }

    v16bf a = frag_global(arow, ks * 32, khalf);
#pragma unroll
    for (int q = 0; q < 3; ++q) {
      // batch the 8 DS reads of this gate's 4 tiles, then 4 WMMAs
      v16bf bfr[4];
#pragma unroll
      for (int t = 0; t < 4; ++t)
        bfr[t] = frag_lds(sbase + cur * bufstride +
                          (q * 64 + t * 16 + row) * ROW_BYTES, khalf);
      SCHED_FENCE();
#pragma unroll
      for (int t = 0; t < 4; ++t)
        acc[q][t] = __builtin_amdgcn_wmma_f32_16x16x32_bf16(
            false, a, false, bfr[t], (short)0, acc[q][t], false, false);
    }
  }

#pragma unroll
  for (int t = 0; t < 4; ++t) {
    const int j = j0 + t * 16 + row;
    const float br = bhh[j];
    const float bz = bhh[HID_ + j];
    const float bn = bhh[2 * HID_ + j];
#pragma unroll
    for (int r = 0; r < 8; ++r) {
      const int b = b0 + khalf * 8 + r;
      const long gi_off = ((long)b * S_DIM + s) * GATES_;
      const float ir  = GI[gi_off + j];
      const float iz  = GI[gi_off + HID_ + j];
      const float inn = GI[gi_off + 2 * HID_ + j];
      const float gr = acc[0][t][r] + br;
      const float gz = acc[1][t][r] + bz;
      const float gn = acc[2][t][r] + bn;
      const float rg = 1.0f / (1.0f + __expf(-(ir + gr)));
      const float zg = 1.0f / (1.0f + __expf(-(iz + gz)));
      const float ng = tanhf(inn + rg * gn);
      const float hp = Hf[(long)b * HID_ + j];
      const float hn = (1.0f - zg) * ng + zg * hp;
      HfN[(long)b * HID_ + j]  = hn;
      HbfN[(long)b * HID_ + j] = (__bf16)hn;
      outH[((long)b * S_DIM + s) * HID_ + j] = hn;
      if (outLast) outLast[(long)b * HID_ + j] = hn;
    }
  }
}

// --- host launcher ------------------------------------------------------
extern "C" void kernel_launch(void* const* d_in, const int* in_sizes, int n_in,
                              void* d_out, int out_size, void* d_ws, size_t ws_size,
                              hipStream_t stream) {
  const float* x   = (const float*)d_in[0];   // [B,S,IN]
  const float* wih = (const float*)d_in[1];   // [3H,IN]
  const float* whh = (const float*)d_in[2];   // [3H,H]
  const float* bih = (const float*)d_in[3];   // [3H]
  const float* bhh = (const float*)d_in[4];   // [3H]
  float* out = (float*)d_out;                 // hiddens [B,S,H] ++ h_last [B,H]

  const long Xe   = (long)B_DIM * S_DIM * IN_DIMS;
  const long WIHe = (long)GATES_ * IN_DIMS;
  const long WHHe = (long)GATES_ * HID_;
  const long He   = (long)B_DIM * HID_;
  const long GIe  = (long)B_DIM * S_DIM * GATES_;

  char* ws = (char*)d_ws;
  size_t off = 0;
  auto take = [&](size_t bytes) -> char* {
    char* p = ws + off;
    off = (off + bytes + 255) & ~(size_t)255;
    return p;
  };

  __bf16* xbf    = (__bf16*)take(Xe * 2);
  __bf16* wihbf  = (__bf16*)take(WIHe * 2);
  __bf16* whhbf  = (__bf16*)take(WHHe * 2);
  __bf16* hbf[2] = { (__bf16*)take(He * 2), (__bf16*)take(He * 2) };
  float*  hf[2]  = { (float*)take(He * 4), (float*)take(He * 4) };
  float*  gi     = (float*)take(GIe * 4);
  (void)ws_size; (void)in_sizes; (void)n_in; (void)out_size;

  const int T = 256;
  f32_to_bf16_kernel<<<(int)((Xe + T - 1) / T), T, 0, stream>>>(x, xbf, Xe);
  f32_to_bf16_kernel<<<(int)((WIHe + T - 1) / T), T, 0, stream>>>(wih, wihbf, WIHe);
  f32_to_bf16_kernel<<<(int)((WHHe + T - 1) / T), T, 0, stream>>>(whh, whhbf, WHHe);
  zero_h_kernel<<<(int)((He + T - 1) / T), T, 0, stream>>>(hbf[0], hf[0], (int)He);

  // Phase 1: big parallel GEMM for input-side gate projections
  dim3 g1(GATES_ / 64, (B_DIM * S_DIM) / 128);
  gi_gemm_kernel<<<g1, 256, 0, stream>>>(xbf, wihbf, bih, gi);

  // Phase 2: 512 serial steps (graph-captured; W_hh bf16 = 6 MB lives in L2)
  float* outLastBase = out + (long)B_DIM * S_DIM * HID_;
  dim3 gs(HID_ / 64, B_DIM / 64);
  for (int s = 0; s < S_DIM; ++s) {
    const int cur = s & 1;
    float* last = (s == S_DIM - 1) ? outLastBase : nullptr;
    gru_step_kernel<<<gs, 128, 0, stream>>>(hbf[cur], hf[cur], whhbf, gi, bhh,
                                            hbf[cur ^ 1], hf[cur ^ 1], out, last, s);
  }
}
